// FeatureAlign_25434796327066
// MI455X (gfx1250) — matrix operand order repaired and checked by reading, not compiled, not run on hardware
//
#include <hip/hip_runtime.h>
#include <hip/hip_fp16.h>

typedef __attribute__((ext_vector_type(16))) _Float16 v16h;
typedef __attribute__((ext_vector_type(4)))  _Float16 v4h;
typedef __attribute__((ext_vector_type(8)))  float    v8f;

#define B_    8
#define CIN   256
#define COUT  256
#define G_    4
#define CPG   64        // CIN / G
#define K9    9         // KH*KW
#define KDIM  2304      // G * CPG * K9
#define NKC   72        // KDIM / 32
#define NPIX  32768     // B*H*W
#define NPAIR 36        // G * K9
#define PIXWG 32        // pixels (N) per workgroup = 2 N-tiles

// ---------------------------------------------------------------------------
// Permuted K (we own both sides): kperm = pair*64 + c, pair = g*9+kk.
// Chunk kc covers pair = kc/2, c in [ (kc&1)*32, +32 ).
//
// Kernel 1: pre-swizzle w_deform (fp32 [Cout][Cin][3][3]) into f16 in exact
// WMMA 16-bit A-fragment order over permuted K:
//   Aswz[kc][mt][lane][j], j=0..15 halves = one v16h per lane.
//   lane 0..15 : M = mt*16+lane,    k_local = (j<8 ? j   : j+8)
//   lane 16..31: M = mt*16+lane-16, k_local = (j<8 ? j+8 : j+16)
// ---------------------------------------------------------------------------
__global__ __launch_bounds__(256)
void swizzle_weights(const float* __restrict__ wdef, _Float16* __restrict__ aswz) {
    int idx = blockIdx.x * 256 + threadIdx.x;          // 72*16*32*16 = 589824
    int j    = idx & 15;
    int lane = (idx >> 4) & 31;
    int mt   = (idx >> 9) & 15;
    int kc   = idx >> 13;
    if (kc >= NKC) return;
    int m      = mt * 16 + (lane & 15);
    int klocal = ((j < 8) ? j : (j + 8)) + ((lane >> 4) ? 8 : 0);
    int kperm  = kc * 32 + klocal;
    int pair   = kperm >> 6;           // g*9 + kk
    int c      = kperm & 63;
    int g      = pair / 9;
    int kk     = pair - g * 9;
    int cin    = g * CPG + c;
    float wv = wdef[((m * CIN + cin) * 3 + (kk / 3)) * 3 + (kk % 3)];
    aswz[idx] = (_Float16)wv;
}

// ---------------------------------------------------------------------------
// Kernel 2: fused offset-einsum + bilinear im2col (LDS) + WMMA GEMM + ReLU.
// One workgroup (8 wave32) per 32-pixel (2 N-tile) slab; each wave owns
// 2 M-tiles x 2 N-tiles = 4 accumulators; A fragments reused across N-tiles.
// ---------------------------------------------------------------------------
__global__ __launch_bounds__(256)
void feature_align_wmma(const float* __restrict__ x,
                        const float* __restrict__ shp,
                        const float* __restrict__ woff,
                        const _Float16* __restrict__ aswz,
                        float* __restrict__ out) {
    // Per (pair, pixel): 4 pre-clamped corner offsets + 4 masked bilinear weights
    __shared__ int   sL00[NPAIR * PIXWG], sL01[NPAIR * PIXWG];
    __shared__ int   sL10[NPAIR * PIXWG], sL11[NPAIR * PIXWG];
    __shared__ float sW00[NPAIR * PIXWG], sW01[NPAIR * PIXWG];
    __shared__ float sW10[NPAIR * PIXWG], sW11[NPAIR * PIXWG];
    __shared__ _Float16 Blds[2][2][32][16];  // [buf][ntile][frag lane][half]

    const int t     = threadIdx.x;
    const int lane  = t & 31;
    const int wave  = t >> 5;
    const int pix0  = blockIdx.x * PIXWG;
    const int b     = pix0 >> 12;            // 32 | 4096: one image per WG

    // ---- Phase 0: offsets -> masked bilinear taps, 36 pairs x 32 pixels
    for (int e = t; e < NPAIR * PIXWG; e += 256) {
        int p    = e & 31;
        int pair = e >> 5;                   // g*9 + kk
        int kk   = pair % 9;
        int hw   = (pix0 + p) & 4095;
        int h    = hw >> 6, w = hw & 63;
        const float* sp = shp + ((size_t)b << 14) + hw;
        float s0 = sp[0], s1 = sp[4096], s2 = sp[8192], s3 = sp[12288];
        const float* wo = woff + pair * 8;
        float dy = wo[0]*s0 + wo[1]*s1 + wo[2]*s2 + wo[3]*s3;
        float dx = wo[4]*s0 + wo[5]*s1 + wo[6]*s2 + wo[7]*s3;
        float py = (float)(kk / 3 - 1 + h) + dy;
        float px = (float)(kk % 3 - 1 + w) + dx;
        float y0f = floorf(py), x0f = floorf(px);
        int   y0 = (int)y0f, x0 = (int)x0f;
        float wy1 = py - y0f, wx1 = px - x0f;
        float wy0 = 1.0f - wy1, wx0 = 1.0f - wx1;
        bool vy0 = (unsigned)y0       < 64u, vy1 = (unsigned)(y0 + 1) < 64u;
        bool vx0 = (unsigned)x0       < 64u, vx1 = (unsigned)(x0 + 1) < 64u;
        int yc0 = min(max(y0,     0), 63) << 6;
        int yc1 = min(max(y0 + 1, 0), 63) << 6;
        int xc0 = min(max(x0,     0), 63);
        int xc1 = min(max(x0 + 1, 0), 63);
        sL00[e] = yc0 + xc0;  sW00[e] = (vy0 && vx0) ? wy0 * wx0 : 0.0f;
        sL01[e] = yc0 + xc1;  sW01[e] = (vy0 && vx1) ? wy0 * wx1 : 0.0f;
        sL10[e] = yc1 + xc0;  sW10[e] = (vy1 && vx0) ? wy1 * wx0 : 0.0f;
        sL11[e] = yc1 + xc1;  sW11[e] = (vy1 && vx1) ? wy1 * wx1 : 0.0f;
    }
    __syncthreads();

    v8f acc00 = {}, acc01 = {}, acc10 = {}, acc11 = {};
    const int mt0 = wave * 2, mt1 = wave * 2 + 1;

    // Fill mapping: thread -> (pixel p5, 4 consecutive k-rows kbase..kbase+3)
    const int p5    = t & 31;                // pixel within slab
    const int kbase = (t >> 5) * 4;          // 0,4,...,28
    const int ntloc = p5 >> 4;               // which N-tile this pixel is in
    const int brow  = (p5 & 15) + ((kbase & 8) << 1);
    const int bslot = (kbase & 7) + ((kbase & 16) >> 1);  // multiple of 4 -> b64 store

    for (int pair = 0; pair < NPAIR; ++pair) {
        const int g = pair / 9;              // uniform per iteration
        const float* plane0 = x + (((size_t)(b * CIN + g * CPG)) << 12);
        const int e = pair * PIXWG + p5;
        const int   l00 = sL00[e], l01 = sL01[e], l10 = sL10[e], l11 = sL11[e];
        const float w00 = sW00[e], w01 = sW01[e], w10 = sW10[e], w11 = sW11[e];

#pragma unroll
        for (int half = 0; half < 2; ++half) {
            const int kc  = pair * 2 + half;
            const int buf = kc & 1;
            // ---- fill: 4 consecutive channels, one pixel, 16 gathers
            const float* xq = plane0 + ((size_t)(half * 32 + kbase) << 12);
            v4h pk;
#pragma unroll
            for (int q = 0; q < 4; ++q) {
                const float* xp = xq + ((size_t)q << 12);
                float v = w00 * xp[l00] + w01 * xp[l01]
                        + w10 * xp[l10] + w11 * xp[l11];
                pk[q] = (_Float16)v;
            }
            *(v4h*)(&Blds[buf][ntloc][brow][bslot]) = pk;   // ds_store_b64
            __syncthreads();

            // ---- consume: 2 A frags x 2 B frags = 4 WMMA per wave
            v16h b0 = *(const v16h*)(&Blds[buf][0][lane][0]);
            v16h b1 = *(const v16h*)(&Blds[buf][1][lane][0]);
            v16h a0 = *(const v16h*)(aswz + ((size_t)((kc*16 + mt0)*32 + lane) << 4));
            v16h a1 = *(const v16h*)(aswz + ((size_t)((kc*16 + mt1)*32 + lane) << 4));
            acc00 = __builtin_amdgcn_wmma_f32_16x16x32_f16(false, a0, false, b0,
                                                           (short)0, acc00, false, false);
            acc01 = __builtin_amdgcn_wmma_f32_16x16x32_f16(false, a0, false, b1,
                                                           (short)0, acc01, false, false);
            acc10 = __builtin_amdgcn_wmma_f32_16x16x32_f16(false, a1, false, b0,
                                                           (short)0, acc10, false, false);
            acc11 = __builtin_amdgcn_wmma_f32_16x16x32_f16(false, a1, false, b1,
                                                           (short)0, acc11, false, false);
        }
    }

    // ---- store + ReLU; C/D layout: lane<16: N=lane, VGPR r -> M=r;
    //      lane>=16: N=lane-16, VGPR r -> M=r+8.
    const int n     = lane & 15;
    const int mbase = (lane >> 4) * 8;
    const int hw0   = (pix0 + n) & 4095;        // N-tile 0
    const int hw1   = (pix0 + 16 + n) & 4095;   // N-tile 1
    float* ob0 = out + (((size_t)b * COUT) << 12) + hw0;
    float* ob1 = out + (((size_t)b * COUT) << 12) + hw1;
#pragma unroll
    for (int r = 0; r < 8; ++r) {
        int m = mbase + r;
        size_t o0 = (size_t)(mt0 * 16 + m) << 12;
        size_t o1 = (size_t)(mt1 * 16 + m) << 12;
        ob0[o0] = fmaxf(acc00[r], 0.0f);
        ob1[o0] = fmaxf(acc01[r], 0.0f);
        ob0[o1] = fmaxf(acc10[r], 0.0f);
        ob1[o1] = fmaxf(acc11[r], 0.0f);
    }
}

extern "C" void kernel_launch(void* const* d_in, const int* in_sizes, int n_in,
                              void* d_out, int out_size, void* d_ws, size_t ws_size,
                              hipStream_t stream) {
    const float* x    = (const float*)d_in[0];   // [8,256,64,64]
    const float* shp  = (const float*)d_in[1];   // [8,4,64,64]
    const float* woff = (const float*)d_in[2];   // [72,4]
    const float* wdef = (const float*)d_in[3];   // [256,256,3,3]
    float* out = (float*)d_out;                  // [8,256,64,64]

    _Float16* aswz = (_Float16*)d_ws;            // 589824 halves = 1.18 MB scratch

    swizzle_weights<<<(NKC * 16 * 32 * 16) / 256, 256, 0, stream>>>(wdef, aswz);
    feature_align_wmma<<<NPIX / PIXWG, 256, 0, stream>>>(x, shp, woff, aswz, out);
}
